// LEIGNN_34376918238022
// MI455X (gfx1250) — matrix operand order repaired and checked by reading, not compiled, not run on hardware
//
#include <hip/hip_runtime.h>
#include <math.h>

#define N_ATOMS 30000
#define N_EDGES 600000
#define HID 128
#define NRBF 128
#define NLAYERS 3
#define NGRAPH 32
#define INV3 0.57735026918962576f
#define INVH 0.08838834764831845f
#define INV2 0.70710678118654752f

typedef __attribute__((ext_vector_type(16))) __bf16 v16bf;
typedef __attribute__((ext_vector_type(8)))  float  v8f;

__device__ __forceinline__ float ssilu_f(float v) {
  return v * (1.0f / (1.0f + expf(-v))) * (1.0f / 0.6f);
}

// ---------------------------------------------------------------------------
// Generic fused GEMM:  out = [resid +] act(A @ Wt^T + bias)
// A: [M,K] fp32 or bf16 (template-selected).  Wt: [Ncol,K] bf16 (pre-transposed).
// Block: 128 threads = 4 waves; block tile = 16 rows x 64 cols (one 16x16
// WMMA tile per wave).  K staged 64 columns at a time into LDS (2 KB), so each
// __syncthreads pair covers TWO v_wmma_f32_16x16x32_bf16 issues.
// Staging: bf16 path = 1x global_load_b128 -> 1x ds_store_b128 per thread;
// fp32 path = 2x global_load_b128 -> 4x v_cvt_pk_bf16_f32 -> 1x ds_store_b128.
// ---------------------------------------------------------------------------
template <bool A_IS_F32>
__global__ void __launch_bounds__(128)
gemm_wmma_bf16(const void* __restrict__ Aptr, const __bf16* __restrict__ Wt,
               const float* __restrict__ bias, const float* __restrict__ resid,
               float* __restrict__ outF, __bf16* __restrict__ outH,
               int M, int K, int Ncol, int act)
{
  __shared__ __bf16 Atile[16 * 64];
  const int tid  = threadIdx.x;
  const int lane = tid & 31;
  const int wv   = tid >> 5;
  const int half = lane >> 4;
  const int ml   = lane & 15;
  const int row0 = blockIdx.x * 16;
  const int ntile = blockIdx.y * 4 + wv;
  const int ncol0 = ntile * 16 + ml;               // column owned by this lane (B/C frag)
  const __bf16* wrow = Wt + (size_t)ncol0 * K;

  // staging coordinates: this thread owns 8 consecutive bf16 of the 16x64 tile
  const int sr = tid >> 3;                         // tile row 0..15
  const int sc = (tid & 7) * 8;                    // tile col 0,8,..,56

  v8f acc = {};
  for (int kk = 0; kk < K; kk += 64) {
    __syncthreads();
    if (A_IS_F32) {
      const float* ap = (const float*)Aptr + (size_t)(row0 + sr) * K + kk + sc;
      const float4 f0 = *(const float4*)ap;
      const float4 f1 = *(const float4*)(ap + 4);
      union { __bf16 h[8]; uint4 u; } pk;
      pk.h[0] = (__bf16)f0.x; pk.h[1] = (__bf16)f0.y;
      pk.h[2] = (__bf16)f0.z; pk.h[3] = (__bf16)f0.w;
      pk.h[4] = (__bf16)f1.x; pk.h[5] = (__bf16)f1.y;
      pk.h[6] = (__bf16)f1.z; pk.h[7] = (__bf16)f1.w;
      *(uint4*)&Atile[sr * 64 + sc] = pk.u;
    } else {
      *(uint4*)&Atile[sr * 64 + sc] =
          *(const uint4*)((const __bf16*)Aptr + (size_t)(row0 + sr) * K + kk + sc);
    }
    __syncthreads();
    #pragma unroll
    for (int ko = 0; ko < 64; ko += 32) {
      v16bf a, b;
      #pragma unroll
      for (int v = 0; v < 8; ++v) {
        // A 16x32 bf16 layout: lanes 0-15 (half=0): v0..3 -> K=0..7, v4..7 -> K=16..23
        //                      lanes16-31 (half=1): v0..3 -> K=8..15, v4..7 -> K=24..31
        int kb = ((v & 4) ? 16 : 0) + half * 8 + (v & 3) * 2;
        a[2 * v]     = Atile[ml * 64 + ko + kb];
        a[2 * v + 1] = Atile[ml * 64 + ko + kb + 1];
        // B 32x16 bf16 layout: same K order as A (half 0: K=0..7,16..23; half 1: K=8..15,24..31)
        int kg = kk + ko + kb;
        b[2 * v]     = wrow[kg];
        b[2 * v + 1] = wrow[kg + 1];
      }
      acc = __builtin_amdgcn_wmma_f32_16x16x32_bf16(false, a, false, b,
                                                    (short)0, acc, false, false);
    }
  }
  float bval = bias ? bias[ncol0] : 0.0f;
  #pragma unroll
  for (int v = 0; v < 8; ++v) {
    int mrow = row0 + v + 8 * half;                 // C/D layout: VGPR v -> M = v + 8*half
    float r = acc[v] + bval;
    if (act) r = ssilu_f(r);
    size_t oi = (size_t)mrow * Ncol + ncol0;
    if (resid) r += resid[oi];
    if (outF) outF[oi] = r;
    if (outH) outH[oi] = (__bf16)r;
  }
}

// W [K,Ncol] fp32 -> Wt [Ncol,K] bf16
__global__ void wt_transpose(const float* __restrict__ W, __bf16* __restrict__ Wt,
                             int K, int Ncol) {
  int idx = blockIdx.x * blockDim.x + threadIdx.x;
  if (idx >= K * Ncol) return;
  int n = idx / K, k = idx - n * K;
  Wt[idx] = (__bf16)W[(size_t)k * Ncol + n];
}

// edge geometry + GemNet radial basis (bf16), one block per edge
__global__ void edge_geom_rbf(const float* __restrict__ pos, const int* __restrict__ ei,
                              float* __restrict__ ev, __bf16* __restrict__ erbf)
{
  int e = blockIdx.x;
  int t = threadIdx.x;                              // 0..127 = rbf index
  int jj = ei[e];
  int ii = ei[N_EDGES + e];
  float dx0 = pos[jj * 3 + 0] - pos[ii * 3 + 0];
  float dy0 = pos[jj * 3 + 1] - pos[ii * 3 + 1];
  float dz0 = pos[jj * 3 + 2] - pos[ii * 3 + 2];
  float dist = sqrtf(dx0 * dx0 + dy0 * dy0 + dz0 * dz0);
  if (t < 3) {
    float comp = (t == 0) ? dx0 : (t == 1 ? dy0 : dz0);
    ev[(size_t)e * 3 + t] = -comp / dist;
  }
  float ds = dist * (1.0f / 6.0f);
  float d5 = ds * ds; d5 *= d5; d5 *= ds;           // ds^5
  float env = 1.0f - 21.0f * d5 + 35.0f * d5 * ds - 15.0f * d5 * ds * ds;
  env = (ds < 1.0f) ? env : 0.0f;
  float off = (float)t * (1.0f / 127.0f);
  float diff = ds - off;
  erbf[(size_t)e * NRBF + t] = (__bf16)(env * expf(-8064.5f * diff * diff));
}

__global__ void init_x(const float* __restrict__ at, const int* __restrict__ z,
                       const int* __restrict__ batch, float* __restrict__ x,
                       float* __restrict__ cnt) {
  int idx = blockIdx.x * blockDim.x + threadIdx.x;
  if (idx >= N_ATOMS * HID) return;
  int n = idx >> 7, h = idx & 127;
  x[idx] = at[(size_t)(z[n] - 1) * HID + h];
  if (h == 0) unsafeAtomicAdd(&cnt[batch[n]], 1.0f);
}

__global__ void init_vx(const float* __restrict__ vnw, float* __restrict__ vx) {
  int idx = blockIdx.x * blockDim.x + threadIdx.x;
  if (idx >= NGRAPH * HID) return;
  vx[idx] = vnw[idx & 127];
}

__global__ void concat_h(const float* __restrict__ x, const float* __restrict__ vx,
                         const int* __restrict__ batch, float* __restrict__ hbuf) {
  int idx = blockIdx.x * blockDim.x + threadIdx.x;
  if (idx >= N_ATOMS * 256) return;
  int n = idx >> 8, c = idx & 255;
  hbuf[idx] = (c < 128) ? x[(size_t)n * 128 + c]
                        : vx[(size_t)batch[n] * 128 + (c - 128)];
}

__global__ void vec_plus_vvec(const float* __restrict__ vec, const float* __restrict__ vvec,
                              const int* __restrict__ batch, float* __restrict__ tmp3) {
  int idx = blockIdx.x * blockDim.x + threadIdx.x;
  if (idx >= N_ATOMS * 3 * HID) return;
  int h = idx & 127, row = idx >> 7;
  int n = row / 3, d = row - n * 3;
  tmp3[idx] = vec[idx] + vvec[((size_t)batch[n] * 3 + d) * HID + h];
}

// message passing: one block per edge, one thread per hidden channel
__global__ void message_k(const int* __restrict__ ei, const __bf16* __restrict__ xh,
                          const __bf16* __restrict__ rbfh, const float* __restrict__ vec,
                          const float* __restrict__ ev, float* __restrict__ dvec,
                          float* __restrict__ dx) {
  int e = blockIdx.x, h = threadIdx.x;
  int jj = ei[e], ii = ei[N_EDGES + e];
  size_t eb = (size_t)e * 384;
  size_t jb = (size_t)jj * 384;
  float x1 = (float)xh[jb + h]       * (float)rbfh[eb + h]       * INV3;
  float x2 = (float)xh[jb + 128 + h] * (float)rbfh[eb + 128 + h] * INV3;
  float x3 = (float)xh[jb + 256 + h] * (float)rbfh[eb + 256 + h] * INV3;
  unsafeAtomicAdd(&dx[(size_t)ii * HID + h], x3);
  #pragma unroll
  for (int d = 0; d < 3; ++d) {
    float vji = (x1 * vec[((size_t)jj * 3 + d) * HID + h] + x2 * ev[(size_t)e * 3 + d]) * INVH;
    unsafeAtomicAdd(&dvec[((size_t)ii * 3 + d) * HID + h], vji);
  }
}

__global__ void apply_msg(float* __restrict__ x, float* __restrict__ vec,
                          const float* __restrict__ dx, const float* __restrict__ dvec) {
  int idx = blockIdx.x * blockDim.x + threadIdx.x;
  if (idx >= N_ATOMS * HID) return;
  x[idx] = (x[idx] + dx[idx]) * INV2;
  int n = idx >> 7, h = idx & 127;
  #pragma unroll
  for (int d = 0; d < 3; ++d) {
    size_t vi = ((size_t)n * 3 + d) * HID + h;
    vec[vi] += dvec[vi];
  }
}

__global__ void vnorm_cat(const float* __restrict__ v12, const float* __restrict__ x,
                          float* __restrict__ cat2) {
  int idx = blockIdx.x * blockDim.x + threadIdx.x;
  if (idx >= N_ATOMS * HID) return;
  int n = idx >> 7, h = idx & 127;
  float s = 1e-8f;
  #pragma unroll
  for (int d = 0; d < 3; ++d) {
    float u = v12[((size_t)n * 3 + d) * 256 + 128 + h];   // vec2 half
    s += u * u;
  }
  cat2[(size_t)n * 256 + h] = x[idx];
  cat2[(size_t)n * 256 + 128 + h] = sqrtf(s);
}

__global__ void update_k(float* __restrict__ x, float* __restrict__ vec,
                         const float* __restrict__ xv, const float* __restrict__ v12) {
  int idx = blockIdx.x * blockDim.x + threadIdx.x;
  if (idx >= N_ATOMS * HID) return;
  int n = idx >> 7, h = idx & 127;
  size_t xb = (size_t)n * 384;
  float a1 = xv[xb + h], a2 = xv[xb + 128 + h], a3 = xv[xb + 256 + h];
  float xx = x[idx];
  x[idx] = xx + a2 * INV2 + xx * tanhf(a3);
  #pragma unroll
  for (int d = 0; d < 3; ++d) {
    size_t vi = ((size_t)n * 3 + d) * HID + h;
    vec[vi] += a1 * v12[((size_t)n * 3 + d) * 256 + h];   // vec1 half
  }
}

__global__ void mean_acc(const float* __restrict__ x, const float* __restrict__ vec,
                         const int* __restrict__ batch, float* __restrict__ xmean,
                         float* __restrict__ vmean) {
  int idx = blockIdx.x * blockDim.x + threadIdx.x;
  if (idx >= N_ATOMS * HID) return;
  int n = idx >> 7, h = idx & 127;
  int b = batch[n];
  unsafeAtomicAdd(&xmean[(size_t)b * HID + h], x[idx]);
  #pragma unroll
  for (int d = 0; d < 3; ++d)
    unsafeAtomicAdd(&vmean[((size_t)b * 3 + d) * HID + h],
                    vec[((size_t)n * 3 + d) * HID + h]);
}

__global__ void build_global(const float* __restrict__ xmean, const float* __restrict__ vmean,
                             const float* __restrict__ vx, const float* __restrict__ vvec,
                             const float* __restrict__ cnt, float* __restrict__ catb,
                             float* __restrict__ tmpB) {
  int idx = blockIdx.x * blockDim.x + threadIdx.x;
  if (idx >= NGRAPH * 3 * HID) return;
  int h = idx & 127, row = idx >> 7;
  int b = row / 3, d = row - b * 3;
  float c = cnt[b];
  tmpB[idx] = vmean[idx] / c + vvec[idx];
  if (d == 0) {
    catb[(size_t)b * 256 + h] = xmean[(size_t)b * HID + h] / c;
    catb[(size_t)b * 256 + 128 + h] = vx[(size_t)b * HID + h];
  }
}

__global__ void energy_k(const float* __restrict__ t, const float* __restrict__ w2,
                         const float* __restrict__ b2, const int* __restrict__ batch,
                         float* __restrict__ out) {
  int n = blockIdx.x * blockDim.x + threadIdx.x;
  if (n >= N_ATOMS) return;
  float s = b2[0];
  #pragma unroll
  for (int k = 0; k < 64; ++k) s += t[(size_t)n * 64 + k] * w2[k];
  unsafeAtomicAdd(&out[batch[n]], s);
}

__global__ void forces_k(const float* __restrict__ vec, const float* __restrict__ wf,
                         float* __restrict__ out) {
  int row = blockIdx.x * blockDim.x + threadIdx.x;
  if (row >= N_ATOMS * 3) return;
  float s = 0.0f;
  #pragma unroll 4
  for (int k = 0; k < 128; ++k) s += vec[(size_t)row * HID + k] * wf[k];
  out[32 + row] = s;
}

extern "C" void kernel_launch(void* const* d_in, const int* in_sizes, int n_in,
                              void* d_out, int out_size, void* d_ws, size_t ws_size,
                              hipStream_t stream) {
  (void)in_sizes; (void)n_in; (void)out_size; (void)ws_size;
  const int N = N_ATOMS, E = N_EDGES, H = HID, B = NGRAPH, L = NLAYERS;

  const float* pos       = (const float*)d_in[0];
  const float* atab      = (const float*)d_in[1];
  const float* vn_w      = (const float*)d_in[2];
  const float* gs_nd_w1  = (const float*)d_in[3];
  const float* gs_nd_b1  = (const float*)d_in[4];
  const float* gs_nd_w2  = (const float*)d_in[5];
  const float* gs_nd_b2  = (const float*)d_in[6];
  const float* gs_vn_w1  = (const float*)d_in[7];
  const float* gs_vn_b1  = (const float*)d_in[8];
  const float* gs_vn_w2  = (const float*)d_in[9];
  const float* gs_vn_b2  = (const float*)d_in[10];
  const float* gv_nd_w   = (const float*)d_in[11];
  const float* gv_vn_w   = (const float*)d_in[12];
  const float* msg_w1    = (const float*)d_in[13];
  const float* msg_b1    = (const float*)d_in[14];
  const float* msg_w2    = (const float*)d_in[15];
  const float* msg_b2    = (const float*)d_in[16];
  const float* rbf_w     = (const float*)d_in[17];
  const float* rbf_b     = (const float*)d_in[18];
  const float* upd_vec_w = (const float*)d_in[19];
  const float* upd_w1    = (const float*)d_in[20];
  const float* upd_b1    = (const float*)d_in[21];
  const float* upd_w2    = (const float*)d_in[22];
  const float* upd_b2    = (const float*)d_in[23];
  const float* out_e_w1  = (const float*)d_in[24];
  const float* out_e_b1  = (const float*)d_in[25];
  const float* out_e_w2  = (const float*)d_in[26];
  const float* out_e_b2  = (const float*)d_in[27];
  const float* out_f_w   = (const float*)d_in[28];
  const int* z     = (const int*)d_in[29];
  const int* batch = (const int*)d_in[30];
  const int* ei    = (const int*)d_in[31];
  float* out = (float*)d_out;

  size_t cur = 0;
  auto carve = [&](size_t bytes) -> char* {
    char* p = (char*)d_ws + cur;
    cur += (bytes + 255) & ~(size_t)255;
    return p;
  };
  float*  ev    = (float*) carve((size_t)E * 3 * 4);
  __bf16* erbf  = (__bf16*)carve((size_t)E * NRBF * 2);
  __bf16* rbfh  = (__bf16*)carve((size_t)E * 384 * 2);
  __bf16* xh16  = (__bf16*)carve((size_t)N * 384 * 2);
  float*  x     = (float*) carve((size_t)N * H * 4);
  float*  vec   = (float*) carve((size_t)N * 3 * H * 4);
  float*  vx    = (float*) carve((size_t)B * H * 4);
  float*  vvec  = (float*) carve((size_t)B * 3 * H * 4);
  float*  cnt   = (float*) carve((size_t)B * 4);
  float*  hbuf  = (float*) carve((size_t)N * 256 * 4);
  float*  t1    = (float*) carve((size_t)N * H * 4);
  float*  dx    = (float*) carve((size_t)N * H * 4);
  float*  dvec  = (float*) carve((size_t)N * 3 * H * 4);
  float*  tmp3  = (float*) carve((size_t)N * 3 * H * 4);
  float*  v12   = (float*) carve((size_t)N * 3 * 256 * 4);
  float*  xv    = (float*) carve((size_t)N * 384 * 4);
  float*  xmean = (float*) carve((size_t)B * H * 4);
  float*  vmean = (float*) carve((size_t)B * 3 * H * 4);
  float*  catb  = (float*) carve((size_t)B * 256 * 4);
  float*  tB    = (float*) carve((size_t)B * H * 4);
  float*  tmpB  = (float*) carve((size_t)B * 3 * H * 4);

  auto carve16 = [&](size_t elems) -> __bf16* { return (__bf16*)carve(elems * 2); };
  auto wtx = [&](const float* W, __bf16* Wt, int K, int Nc) {
    int tot = K * Nc;
    wt_transpose<<<(tot + 255) / 256, 256, 0, stream>>>(W, Wt, K, Nc);
  };

  __bf16 *w_gs_nd_w1[3], *w_gs_nd_w2[3], *w_gv_nd_w[3], *w_msg_w1[3], *w_msg_w2[3],
         *w_rbf_w[3], *w_upd_vec_w[3], *w_upd_w1[3], *w_upd_w2[3],
         *w_gs_vn_w1[3], *w_gs_vn_w2[3], *w_gv_vn_w[3];
  __bf16* w_out_e_w1 = carve16(128 * 64);
  for (int l = 0; l < L; ++l) {
    w_gs_nd_w1[l]  = carve16(256 * 128);
    w_gs_nd_w2[l]  = carve16(128 * 128);
    w_gv_nd_w[l]   = carve16(128 * 128);
    w_msg_w1[l]    = carve16(128 * 64);
    w_msg_w2[l]    = carve16(64 * 384);
    w_rbf_w[l]     = carve16(128 * 384);
    w_upd_vec_w[l] = carve16(128 * 256);
    w_upd_w1[l]    = carve16(256 * 128);
    w_upd_w2[l]    = carve16(128 * 384);
    w_gs_vn_w1[l]  = carve16(256 * 128);
    w_gs_vn_w2[l]  = carve16(128 * 128);
    w_gv_vn_w[l]   = carve16(128 * 128);
    wtx(gs_nd_w1  + (size_t)l * 256 * 128, w_gs_nd_w1[l], 256, 128);
    wtx(gs_nd_w2  + (size_t)l * 128 * 128, w_gs_nd_w2[l], 128, 128);
    wtx(gv_nd_w   + (size_t)l * 128 * 128, w_gv_nd_w[l], 128, 128);
    wtx(msg_w1    + (size_t)l * 128 * 64,  w_msg_w1[l], 128, 64);
    wtx(msg_w2    + (size_t)l * 64 * 384,  w_msg_w2[l], 64, 384);
    wtx(rbf_w     + (size_t)l * 128 * 384, w_rbf_w[l], 128, 384);
    wtx(upd_vec_w + (size_t)l * 128 * 256, w_upd_vec_w[l], 128, 256);
    wtx(upd_w1    + (size_t)l * 256 * 128, w_upd_w1[l], 256, 128);
    wtx(upd_w2    + (size_t)l * 128 * 384, w_upd_w2[l], 128, 384);
    wtx(gs_vn_w1  + (size_t)l * 256 * 128, w_gs_vn_w1[l], 256, 128);
    wtx(gs_vn_w2  + (size_t)l * 128 * 128, w_gs_vn_w2[l], 128, 128);
    wtx(gv_vn_w   + (size_t)l * 128 * 128, w_gv_vn_w[l], 128, 128);
  }
  wtx(out_e_w1, w_out_e_w1, 128, 64);

  auto gemmF = [&](const float* Af, const __bf16* Wt, const float* bias,
                   const float* resid, float* oF, __bf16* oH, int M, int K, int Nc, int act) {
    dim3 g(M / 16, Nc / 64);
    gemm_wmma_bf16<true><<<g, 128, 0, stream>>>((const void*)Af, Wt, bias, resid, oF, oH, M, K, Nc, act);
  };
  auto gemmH = [&](const __bf16* Ah, const __bf16* Wt, const float* bias,
                   const float* resid, float* oF, __bf16* oH, int M, int K, int Nc, int act) {
    dim3 g(M / 16, Nc / 64);
    gemm_wmma_bf16<false><<<g, 128, 0, stream>>>((const void*)Ah, Wt, bias, resid, oF, oH, M, K, Nc, act);
  };

  // ---- init ----
  hipMemsetAsync(vec,  0, (size_t)N * 3 * H * 4, stream);
  hipMemsetAsync(vvec, 0, (size_t)B * 3 * H * 4, stream);
  hipMemsetAsync(cnt,  0, (size_t)B * 4, stream);
  init_x<<<(N * H + 255) / 256, 256, 0, stream>>>(atab, z, batch, x, cnt);
  init_vx<<<(B * H + 255) / 256, 256, 0, stream>>>(vn_w, vx);
  edge_geom_rbf<<<E, 128, 0, stream>>>(pos, ei, ev, erbf);

  // ---- layers ----
  for (int l = 0; l < L; ++l) {
    // GlobalScalar.update_local_emb
    concat_h<<<(N * 256 + 255) / 256, 256, 0, stream>>>(x, vx, batch, hbuf);
    gemmF(hbuf, w_gs_nd_w1[l], gs_nd_b1 + l * H, nullptr, t1, nullptr, N, 256, 128, 1);
    gemmF(t1,   w_gs_nd_w2[l], gs_nd_b2 + l * H, x, x, nullptr, N, 128, 128, 1);
    // GlobalVector.update_local_emb
    vec_plus_vvec<<<(N * 3 * H + 255) / 256, 256, 0, stream>>>(vec, vvec, batch, tmp3);
    gemmF(tmp3, w_gv_nd_w[l], nullptr, vec, vec, nullptr, 3 * N, 128, 128, 0);
    // Message
    gemmH(erbf, w_rbf_w[l], rbf_b + l * 384, nullptr, nullptr, rbfh, E, 128, 384, 0);
    gemmF(x,  w_msg_w1[l], msg_b1 + l * 64, nullptr, t1, nullptr, N, 128, 64, 1);
    gemmF(t1, w_msg_w2[l], msg_b2 + l * 384, nullptr, nullptr, xh16, N, 64, 384, 0);
    hipMemsetAsync(dx,   0, (size_t)N * H * 4, stream);
    hipMemsetAsync(dvec, 0, (size_t)N * 3 * H * 4, stream);
    message_k<<<E, 128, 0, stream>>>(ei, xh16, rbfh, vec, ev, dvec, dx);
    apply_msg<<<(N * H + 255) / 256, 256, 0, stream>>>(x, vec, dx, dvec);
    // Update
    gemmF(vec, w_upd_vec_w[l], nullptr, nullptr, v12, nullptr, 3 * N, 128, 256, 0);
    vnorm_cat<<<(N * H + 255) / 256, 256, 0, stream>>>(v12, x, hbuf);
    gemmF(hbuf, w_upd_w1[l], upd_b1 + l * H, nullptr, t1, nullptr, N, 256, 128, 1);
    gemmF(t1, w_upd_w2[l], upd_b2 + l * 384, nullptr, xv, nullptr, N, 128, 384, 0);
    update_k<<<(N * H + 255) / 256, 256, 0, stream>>>(x, vec, xv, v12);
    // Global updates
    hipMemsetAsync(xmean, 0, (size_t)B * H * 4, stream);
    hipMemsetAsync(vmean, 0, (size_t)B * 3 * H * 4, stream);
    mean_acc<<<(N * H + 255) / 256, 256, 0, stream>>>(x, vec, batch, xmean, vmean);
    build_global<<<(B * 3 * H + 127) / 128, 128, 0, stream>>>(xmean, vmean, vx, vvec, cnt, catb, tmpB);
    gemmF(catb, w_gs_vn_w1[l], gs_vn_b1 + l * H, nullptr, tB, nullptr, B, 256, 128, 1);
    gemmF(tB,   w_gs_vn_w2[l], gs_vn_b2 + l * H, vx, vx, nullptr, B, 128, 128, 1);
    gemmF(tmpB, w_gv_vn_w[l], nullptr, vvec, vvec, nullptr, 3 * B, 128, 128, 0);
  }

  // ---- outputs ----
  gemmF(x, w_out_e_w1, out_e_b1, nullptr, t1, nullptr, N, 128, 64, 1);
  hipMemsetAsync(out, 0, 32 * 4, stream);
  energy_k<<<(N + 255) / 256, 256, 0, stream>>>(t1, out_e_w2, out_e_b2, batch, out);
  forces_k<<<(N * 3 + 255) / 256, 256, 0, stream>>>(vec, out_f_w, out);
}
